// SelectionStatistics_7705171329191
// MI455X (gfx1250) — matrix operand order repaired and checked by reading, not compiled, not run on hardware
//
#include <hip/hip_runtime.h>

#define EPSF 1e-6f

constexpr int Bn   = 8;
constexpr int Tn   = 4096;
constexpr int DUn  = 128;
constexpr int TILE = 128;
constexpr int NT   = Tn / TILE;            // 32 tiles per dim
constexpr int NPAIRS = NT * (NT + 1) / 2;  // 528 upper-triangular tile pairs
constexpr int LDSTRIDE = DUn + 8;          // 136 bf16 (272B rows, 16B aligned, bank-spread)

typedef __attribute__((ext_vector_type(16))) __bf16 v16bf;
typedef __attribute__((ext_vector_type(8)))  __bf16 v8bf;
typedef __attribute__((ext_vector_type(8)))  float  v8f;
typedef __attribute__((ext_vector_type(4)))  float  v4f;

// ---------------------------------------------------------------------------
// Kernel 1: coverage / entropy / spacing per batch; zero compactness accum.
// ---------------------------------------------------------------------------
__global__ __launch_bounds__(256)
void stats_kernel(const float* __restrict__ y, float* __restrict__ out,
                  float* __restrict__ ws) {
    const int b = blockIdx.x;
    const int tid = threadIdx.x;
    __shared__ float s0[256], s1[256], s2[256];

    float a0 = 0.f, a1 = 0.f, a2 = 0.f;
    for (int t = tid; t < Tn; t += 256) {
        float v = y[b * Tn + t];
        float p = (float)t;
        a0 += v; a1 += v * p; a2 += v * p * p;
    }
    s0[tid] = a0; s1[tid] = a1; s2[tid] = a2;
    __syncthreads();
    for (int s = 128; s > 0; s >>= 1) {
        if (tid < s) { s0[tid] += s0[tid+s]; s1[tid] += s1[tid+s]; s2[tid] += s2[tid+s]; }
        __syncthreads();
    }
    const float S0 = s0[0], S1 = s1[0], S2 = s2[0];
    const float tw = fmaxf(S0, EPSF);
    __syncthreads();

    // entropy pass: -sum y_norm * log(clip(y_norm, eps))
    float e = 0.f;
    for (int t = tid; t < Tn; t += 256) {
        float yn = y[b * Tn + t] / tw;
        e -= yn * __logf(fmaxf(yn, EPSF));
    }
    s0[tid] = e;
    __syncthreads();
    for (int s = 128; s > 0; s >>= 1) {
        if (tid < s) s0[tid] += s0[tid+s];
        __syncthreads();
    }

    if (tid == 0) {
        out[b * 4 + 0] = S0 / (float)Tn;                     // coverage
        out[b * 4 + 1] = s0[0];                              // entropy
        float m   = S1 / tw;
        float var = (S2 - 2.f * m * S1 + m * m * S0) / tw;   // exact weighted var
        out[b * 4 + 2] = sqrtf(fmaxf(var, EPSF));            // spacing
        ws[8 + b] = tw;      // stash tw for finalize
        ws[b]     = 0.f;     // zero compactness accumulator (runs before tiles)
    }
}

// ---------------------------------------------------------------------------
// Kernel 2: sq[b][t] = ||u[b,t,:]||^2   (one wave32 per row)
// ---------------------------------------------------------------------------
__global__ __launch_bounds__(256)
void sq_kernel(const float* __restrict__ u, float* __restrict__ sq) {
    const int row  = blockIdx.x * 8 + (threadIdx.x >> 5);   // 8 waves per block
    const int lane = threadIdx.x & 31;
    const float* p = u + (size_t)row * DUn;
    float s = 0.f;
    #pragma unroll
    for (int k = 0; k < DUn; k += 32) { float v = p[k + lane]; s = fmaf(v, v, s); }
    #pragma unroll
    for (int off = 16; off > 0; off >>= 1) s += __shfl_down(s, off, 32);
    if (lane == 0) sq[row] = s;
}

// ---------------------------------------------------------------------------
// Kernel 3: WMMA tile kernel — fused Gram + distance + weighted reduction.
// One block per (upper-triangular 128x128 tile pair, batch).
// ---------------------------------------------------------------------------
__global__ __launch_bounds__(256)
void tile_kernel(const float* __restrict__ u, const float* __restrict__ y,
                 const float* __restrict__ sq, float* __restrict__ accum) {
    const int b   = blockIdx.y;
    const int tid = threadIdx.x;

    // decode upper-triangular pair index -> (it, jt), it <= jt (uniform scalar)
    int it = 0, rem = blockIdx.x;
    while (rem >= NT - it) { rem -= NT - it; ++it; }
    const int jt = it + rem;
    const int i0 = it * TILE, j0 = jt * TILE;

    extern __shared__ char smem[];
    __bf16* As  = (__bf16*)smem;                 // [128][LDSTRIDE] row-major (I tile)
    __bf16* Bs  = As + TILE * LDSTRIDE;          // [128][LDSTRIDE] row-major (J tile)
    float*  yIs = (float*)(Bs + TILE * LDSTRIDE);
    float*  sIs = yIs + TILE;
    float*  yJs = sIs + TILE;
    float*  sJs = yJs + TILE;
    float*  red = sJs + TILE;                    // [256]

    // ---- stage fp32 -> bf16 tiles into LDS (coalesced float4 reads) ----
    const float* uI = u + ((size_t)b * Tn + i0) * DUn;
    const float* uJ = u + ((size_t)b * Tn + j0) * DUn;
    #pragma unroll
    for (int iter = 0; iter < 16; ++iter) {
        int lin = iter * 256 + tid;      // 0..4095
        int r   = lin >> 5;              // row 0..127
        int c   = (lin & 31) << 2;       // k 0..124 step 4
        v4f a4 = *(const v4f*)(uI + r * DUn + c);
        v4f b4 = *(const v4f*)(uJ + r * DUn + c);
        int o = r * LDSTRIDE + c;
        As[o+0] = (__bf16)a4[0]; As[o+1] = (__bf16)a4[1];
        As[o+2] = (__bf16)a4[2]; As[o+3] = (__bf16)a4[3];
        Bs[o+0] = (__bf16)b4[0]; Bs[o+1] = (__bf16)b4[1];
        Bs[o+2] = (__bf16)b4[2]; Bs[o+3] = (__bf16)b4[3];
    }
    if (tid < 128) {
        yIs[tid] = y [(size_t)b * Tn + i0 + tid];
        sIs[tid] = sq[(size_t)b * Tn + i0 + tid];
    } else {
        int t2 = tid - 128;
        yJs[t2] = y [(size_t)b * Tn + j0 + t2];
        sJs[t2] = sq[(size_t)b * Tn + j0 + t2];
    }
    __syncthreads();

    // ---- per-wave WMMA: wave owns 16-row band of the 128x128 output tile ----
    const int lane = tid & 31, wave = tid >> 5;
    const int l15  = lane & 15, half = lane >> 4;

    // A fragments: lane m (0-15) = row, K chunks {kc*32 + 8*half + [0..7], +16}
    v16bf afrag[4];
    const __bf16* abase = As + (wave * 16 + l15) * LDSTRIDE;
    #pragma unroll
    for (int kc = 0; kc < 4; ++kc) {
        const __bf16* p = abase + kc * 32 + half * 8;
        v8bf lo = *(const v8bf*)p;
        v8bf hi = *(const v8bf*)(p + 16);
        afrag[kc] = __builtin_shufflevector(lo, hi,
            0,1,2,3,4,5,6,7,8,9,10,11,12,13,14,15);
    }

    // per-lane row weights for the C/D layout (VGPR r -> M = r + 8*half)
    float yi[8], si[8];
    #pragma unroll
    for (int r = 0; r < 8; ++r) {
        int ii = wave * 16 + half * 8 + r;
        yi[r] = yIs[ii]; si[r] = sIs[ii];
    }

    float partial = 0.f;
    #pragma unroll
    for (int jtile = 0; jtile < 8; ++jtile) {
        const int n0 = jtile * 16;
        const float yj = yJs[n0 + l15];
        const float sj = sJs[n0 + l15];

        v8f acc = {0.f, 0.f, 0.f, 0.f, 0.f, 0.f, 0.f, 0.f};
        #pragma unroll
        for (int kc = 0; kc < 4; ++kc) {
            // B fragment: lane n (0-15) = column -> row (j0+n0+n) of U,
            // K chunks {kc*32 + 8*half + [0..7], +16}
            const __bf16* q = Bs + (n0 + l15) * LDSTRIDE + kc * 32 + half * 8;
            v8bf lo = *(const v8bf*)q;
            v8bf hi = *(const v8bf*)(q + 16);
            v16bf bfrag = __builtin_shufflevector(lo, hi,
                0,1,2,3,4,5,6,7,8,9,10,11,12,13,14,15);
            acc = __builtin_amdgcn_wmma_f32_16x16x32_bf16(
                false, afrag[kc], false, bfrag, (short)0, acc, false, false);
        }

        // fused epilogue: dist = sqrt(max(sq_i+sq_j-2g,0)+eps); w = y_i*y_j
        #pragma unroll
        for (int r = 0; r < 8; ++r) {
            float d2 = fmaxf(si[r] + sj - 2.0f * acc[r], 0.0f);
            partial  = fmaf(yi[r] * yj, __fsqrt_rn(d2 + EPSF), partial);
        }
    }

    // ---- block reduce + atomic accumulate (x2 for off-diagonal tile pairs) ----
    red[tid] = partial;
    __syncthreads();
    for (int s = 128; s > 0; s >>= 1) {
        if (tid < s) red[tid] += red[tid + s];
        __syncthreads();
    }
    if (tid == 0) {
        float v = red[0] * ((it != jt) ? 2.0f : 1.0f);
        atomicAdd(&accum[b], v);
    }
}

// ---------------------------------------------------------------------------
// Kernel 4: compactness = accum / clip(tw^2, eps)
// ---------------------------------------------------------------------------
__global__ void finalize_kernel(float* __restrict__ out, const float* __restrict__ ws) {
    int b = threadIdx.x;
    if (b < Bn) {
        float tw = ws[8 + b];
        out[b * 4 + 3] = ws[b] / fmaxf(tw * tw, EPSF);
    }
}

// ---------------------------------------------------------------------------
extern "C" void kernel_launch(void* const* d_in, const int* in_sizes, int n_in,
                              void* d_out, int out_size, void* d_ws, size_t ws_size,
                              hipStream_t stream) {
    const float* y = (const float*)d_in[0];   // [B,T]
    const float* u = (const float*)d_in[1];   // [B,T,DU]
    float* out = (float*)d_out;               // [B,4]
    float* ws  = (float*)d_ws;
    float* accum = ws;            // [8]  compactness accumulators
    float* sq    = ws + 16;       // [B*T] row squared norms

    stats_kernel<<<Bn, 256, 0, stream>>>(y, out, ws);
    sq_kernel<<<(Bn * Tn) / 8, 256, 0, stream>>>(u, sq);

    const size_t smem = (size_t)2 * TILE * LDSTRIDE * sizeof(__bf16)  // A,B tiles
                      + 4 * TILE * sizeof(float)                      // y/sq stage
                      + 256 * sizeof(float);                          // reduction
    tile_kernel<<<dim3(NPAIRS, Bn), 256, smem, stream>>>(u, y, sq, accum);

    finalize_kernel<<<1, Bn, 0, stream>>>(out, ws);
}
